// EnsembleRSSM_33457795236002
// MI455X (gfx1250) — matrix operand order repaired and checked by reading, not compiled, not run on hardware
//
#include <hip/hip_runtime.h>
#include <hip/hip_bf16.h>
#include <stdint.h>

// Problem sizes (fixed by the reference)
#define TT   64
#define BB   512
#define OBS_ 1536
#define ACT_ 32
#define DD   2048
#define HH   2048
#define KK_  32
#define VV_  32
#define SS   1024

typedef __attribute__((ext_vector_type(8)))  unsigned short us8;
typedef __attribute__((ext_vector_type(16))) unsigned short us16;
typedef __attribute__((ext_vector_type(16))) __bf16        v16bf;
typedef __attribute__((ext_vector_type(8)))  float         v8f;
typedef __attribute__((ext_vector_type(4)))  unsigned int  u32x4;
typedef __attribute__((ext_vector_type(8)))  int           i32x8;
typedef __attribute__((ext_vector_type(4)))  int           i32x4;

#if defined(__has_builtin)
#if __has_builtin(__builtin_amdgcn_tensor_load_to_lds) && \
    __has_builtin(__builtin_amdgcn_s_wait_tensorcnt)
#define USE_TDM 1
#endif
#endif
#ifndef USE_TDM
#define USE_TDM 0
#endif

// Native fp32->bf16 (v_cvt_pk_bf16_f32 confirmed in round-2 asm)
__device__ __forceinline__ unsigned short f2bf(float f) {
  return __builtin_bit_cast(unsigned short, static_cast<__bf16>(f));
}

#if USE_TDM
// ---------------------------------------------------------------------------
// TDM: one descriptor loads a whole contiguous 8KB B tile (4096 bf16) from
// global into LDS, inserting a 16B pad after every 64B of data -- exactly the
// padded [n][k] layout (row = 32 bf16 data + 8 bf16 pad = LDB_S 40).
//   D# group0: count=1 | lds_addr | global_addr | type=2
//   D# group1: data_size=2B, pad_enable, pad_interval=3 (16 DW = 64B),
//              pad_amount=3 (4 DW = 16B), tensor_dim0=tile_dim0=4096,
//              tensor_dim1=1, tensor_dim0_stride=4096
//   groups 2/3: zero (<=2D tensor)
// This toolchain's builtin takes the 6-arg form:
//   (uint32x4 g0, int32x8 g1, int32x4, int32x4, int32x8, i32 cpol)
// ---------------------------------------------------------------------------
__device__ __forceinline__ void tdm_load_b_tile(const unsigned short* gsrc,
                                                unsigned lds_off) {
  const unsigned long long ga = (unsigned long long)(uintptr_t)gsrc;
  u32x4 g0;
  g0[0] = 1u;                                   // count=1 (valid descriptor)
  g0[1] = lds_off;                              // LDS byte address
  g0[2] = (unsigned)(ga & 0xFFFFFFFFu);         // global_addr[31:0]
  g0[3] = (unsigned)((ga >> 32) & 0x01FFFFFFu)  // global_addr[56:32]
        | (2u << 30);                           // type = 2 ("image")
  i32x8 g1;
  g1[0] = (1 << 16)                             // data_size = 2 bytes
        | (1 << 20)                             // pad_enable
        | (3 << 22)                             // pad_interval: 16 DWORDs
        | (3 << 25);                            // pad_amount:   4 DWORDs
  g1[1] = (int)(4096u << 16);                   // tensor_dim0 lo16 -> [31:16]
  g1[2] = (int)(1u << 16);                      // tensor_dim1 = 1
  g1[3] = (int)(4096u << 16);                   // tile_dim0 = 4096 elems
  g1[4] = 0;                                    // tile_dim1/2 unused
  g1[5] = 4096;                                 // tensor_dim0_stride lo32
  g1[6] = 0;
  g1[7] = 0;
  const i32x4 gz4 = {0, 0, 0, 0};
  const i32x8 gz8 = {0, 0, 0, 0, 0, 0, 0, 0};
  __builtin_amdgcn_tensor_load_to_lds(g0, g1, gz4, gz4, gz8, 0);
}
#endif

// ---------------------------------------------------------------------------
// Weight pack: fp32 [K][N] row-major -> bf16 tile-swizzled
//   dst[nt][kt][n][k],  tile = 128 (n) x 32 (k),  tile stride = 4096 elems.
// ktOff / kTilesTot let two sources stack along K into one packed matrix
// (e.g. [W_ia; W_is]). Source reads are fully coalesced (linear in src).
// ---------------------------------------------------------------------------
__global__ void pack_w_kernel(const float* __restrict__ src,
                              unsigned short* __restrict__ dst,
                              int N, int Ksrc, int ktOff, int kTilesTot) {
  const size_t total = (size_t)Ksrc * N;
  for (size_t i = (size_t)blockIdx.x * blockDim.x + threadIdx.x; i < total;
       i += (size_t)gridDim.x * blockDim.x) {
    const int srck = (int)(i / N);
    const int srcn = (int)(i % N);
    const int nt = srcn >> 7, n = srcn & 127;
    const int kt = ktOff + (srck >> 5), k = srck & 31;
    dst[(((size_t)nt * kTilesTot + kt) << 12) + ((size_t)n << 5) + k] =
        f2bf(src[i]);
  }
}

__global__ void add2_kernel(const float* __restrict__ a,
                            const float* __restrict__ b,
                            float* __restrict__ o, int n) {
  int i = blockIdx.x * blockDim.x + threadIdx.x;
  if (i < n) o[i] = a[i] + b[i];
}

// ---------------------------------------------------------------------------
// bf16 WMMA GEMM:  C[M,N] = act( [A0 | A1][M,Ktot] * B[Ktot,N] + bias )
//   A: fp32 global (two row-major segments split at K0), cvt to bf16 while
//      staging into LDS [m][k].
//   B: pre-swizzled bf16 tiles [nt][kt][128][32]; per K-tile loaded by ONE
//      TDM tensor_load_to_lds (wave 0 issues; hw applies LDS padding), or by
//      the register-staged fallback path if the builtin is unavailable.
//   Block tile 64x128, BK=32, double-buffered LDS; 8 waves (2x4),
//   wave tile 32x32 -> 2x2 accumulators, 4 WMMAs per K-tile per wave.
//   act: 0 = none, 1 = elu
// Assumes: M%64==0, N%128==0, Ktot%32==0, K0%32==0.
// ---------------------------------------------------------------------------
#define BM 64
#define BN 128
#define BK 32
#define LDA_S 40   // padded LDS stride (bf16 elems) for A tile [m][k]
#define LDB_S 40   // padded LDS stride for B tile [n][k] (= 64B data + 16B pad)

__global__ __launch_bounds__(256) void gemm_bf16_wmma(
    const float* __restrict__ A0, int lda0, int K0,
    const float* __restrict__ A1, int lda1,
    const unsigned short* __restrict__ Bp,   // packed tiles
    int N, int Ktot, const float* __restrict__ bias,
    float* __restrict__ C, int ldc, int act)
{
  __shared__ unsigned short ldsA[2][BM * LDA_S];
  __shared__ unsigned short ldsB[2][BN * LDB_S];

  const int tid   = threadIdx.x;
  const int lane  = tid & 31;
  const int wid   = tid >> 5;          // 0..7
  const int waveM = wid >> 2;          // 0..1 -> 32 rows each
  const int waveN = wid & 3;           // 0..3 -> 32 cols each
  const int lrow  = lane & 15;
  const int lhi   = lane >> 4;         // 0 or 1

  const int bm = blockIdx.y * BM;
  const int bn = blockIdx.x;           // tile index along N
  const int kTiles = Ktot >> 5;

  v8f acc[2][2];
  const v8f vzero = {0.f,0.f,0.f,0.f,0.f,0.f,0.f,0.f};
  #pragma unroll
  for (int i = 0; i < 2; ++i)
    #pragma unroll
    for (int j = 0; j < 2; ++j) acc[i][j] = vzero;

  // staging coordinates
  const int ar = tid >> 2;             // A row in tile (0..63)
  const int ac = (tid & 3) * 8;        // A k sub-offset (0,8,16,24)

  float4 a4[2];
  auto fetchA = [&](int kt) {
    const int kg = kt * BK + ac;
    const float* src = (kg < K0)
        ? A0 + (size_t)(bm + ar) * lda0 + kg
        : A1 + (size_t)(bm + ar) * lda1 + (kg - K0);
    const float4* s4 = (const float4*)src;
    a4[0] = s4[0]; a4[1] = s4[1];
  };
  auto stageA = [&](int buf) {
    us8 x;
    x[0] = f2bf(a4[0].x); x[1] = f2bf(a4[0].y);
    x[2] = f2bf(a4[0].z); x[3] = f2bf(a4[0].w);
    x[4] = f2bf(a4[1].x); x[5] = f2bf(a4[1].y);
    x[6] = f2bf(a4[1].z); x[7] = f2bf(a4[1].w);
    *(us8*)&ldsA[buf][ar * LDA_S + ac] = x;        // 16B aligned (80r+2c)
  };

#if USE_TDM
  const bool issuer = (wid == 0);      // one TDM issue per workgroup
  auto loadB = [&](int kt, int buf) {
    if (issuer)
      tdm_load_b_tile(Bp + (((size_t)bn * kTiles + kt) << 12),
                      (unsigned)(uintptr_t)&ldsB[buf][0]);
  };
#else
  const int br = tid >> 1;             // B n in tile (0..127)
  const int bc = (tid & 1) * 16;       // B k sub-offset (0 or 16)
  us8 b0, b1;
  auto fetchB = [&](int kt) {
    const us8* bsrc = (const us8*)(Bp + (((size_t)bn * kTiles + kt) << 12));
    b0 = bsrc[2 * tid]; b1 = bsrc[2 * tid + 1];
  };
  auto stageB = [&](int buf) {
    *(us8*)&ldsB[buf][br * LDB_S + bc + 0] = b0;
    *(us8*)&ldsB[buf][br * LDB_S + bc + 8] = b1;
  };
#endif

  // ---- prologue: tile 0 ----
  fetchA(0);
#if USE_TDM
  loadB(0, 0);
#else
  fetchB(0);
#endif
  stageA(0);
#if USE_TDM
  if (issuer) __builtin_amdgcn_s_wait_tensorcnt(0);
#else
  stageB(0);
#endif
  __syncthreads();

  for (int kt = 0; kt < kTiles; ++kt) {
    const int cur  = kt & 1;
    const bool more = (kt + 1 < kTiles);
    if (more) {
      fetchA(kt + 1);
#if USE_TDM
      loadB(kt + 1, cur ^ 1);
#else
      fetchB(kt + 1);
      if (kt + 2 < kTiles)             // -> global_prefetch_b8
        __builtin_prefetch(
            Bp + (((size_t)bn * kTiles + kt + 2) << 12) + (size_t)tid * 16,
            0, 1);
#endif
    }

    // ---- fragments + WMMA from lds[cur] ----
    // A frag (16x32): lane<16 holds M=lrow, K {0..7,16..23};
    //                 lane>=16 same M, K {8..15,24..31}.
    v16bf afrag[2];
    #pragma unroll
    for (int fm = 0; fm < 2; ++fm) {
      const unsigned short* pa =
          &ldsA[cur][(waveM * 32 + fm * 16 + lrow) * LDA_S];
      us8 lo = *(const us8*)(pa + lhi * 8);
      us8 hi = *(const us8*)(pa + 16 + lhi * 8);
      us16 all = __builtin_shufflevector(lo, hi, 0,1,2,3,4,5,6,7,
                                                 8,9,10,11,12,13,14,15);
      afrag[fm] = __builtin_bit_cast(v16bf, all);
    }
    // B frag (32x16): lane holds N=lrow, K {lhi*8..+7, 16+lhi*8..+7} (same as A).
    #pragma unroll
    for (int fn = 0; fn < 2; ++fn) {
      const unsigned short* pb =
          &ldsB[cur][(waveN * 32 + fn * 16 + lrow) * LDB_S + lhi * 8];
      us8 lo = *(const us8*)(pb);
      us8 hi = *(const us8*)(pb + 16);
      us16 all = __builtin_shufflevector(lo, hi, 0,1,2,3,4,5,6,7,
                                                 8,9,10,11,12,13,14,15);
      v16bf bfrag = __builtin_bit_cast(v16bf, all);
      #pragma unroll
      for (int fm = 0; fm < 2; ++fm) {
        acc[fm][fn] = __builtin_amdgcn_wmma_f32_16x16x32_bf16(
            false, afrag[fm], false, bfrag, (short)0, acc[fm][fn],
            false, false);
      }
    }

    if (more) {
      stageA(cur ^ 1);
#if !USE_TDM
      stageB(cur ^ 1);
#endif
    }
#if USE_TDM
    if (issuer && more) __builtin_amdgcn_s_wait_tensorcnt(0);
#endif
    __syncthreads();
  }

  // ---- epilogue: bias + activation, fp32 store ----
  // C/D layout: VGPR r -> M = r + 8*lhi ; N = lrow (within 16x16 tile).
  #pragma unroll
  for (int fn = 0; fn < 2; ++fn) {
    const int n = bn * BN + waveN * 32 + fn * 16 + lrow;
    const float bv = bias ? bias[n] : 0.f;
    #pragma unroll
    for (int fm = 0; fm < 2; ++fm) {
      #pragma unroll
      for (int r = 0; r < 8; ++r) {
        const int m = bm + waveM * 32 + fm * 16 + r + lhi * 8;
        float v = acc[fm][fn][r] + bv;
        if (act == 1) v = (v > 0.f) ? v : (expf(v) - 1.f);
        C[(size_t)m * ldc + n] = v;
      }
    }
  }
}

// ---------------------------------------------------------------------------
// LayerNorm(6144) + GRU gate update. One block (256 thr) per batch row.
// ---------------------------------------------------------------------------
__global__ __launch_bounds__(256) void gru_ln_kernel(
    const float* __restrict__ P, const float* __restrict__ ln_g,
    const float* __restrict__ ln_b, float* __restrict__ deter_state,
    float* __restrict__ deters_out)
{
  __shared__ float red[16];
  const int b = blockIdx.x;
  const int tid = threadIdx.x;
  const float* p = P + (size_t)b * 3 * DD;

  float s = 0.f, s2 = 0.f;
  for (int i = tid; i < 3 * DD; i += 256) {
    float v = p[i]; s += v; s2 += v * v;
  }
  for (int off = 16; off > 0; off >>= 1) {
    s  += __shfl_down(s,  off, 32);
    s2 += __shfl_down(s2, off, 32);
  }
  if ((tid & 31) == 0) { red[(tid >> 5) * 2] = s; red[(tid >> 5) * 2 + 1] = s2; }
  __syncthreads();
  if (tid == 0) {
    float S = 0.f, S2 = 0.f;
    for (int w2 = 0; w2 < 8; ++w2) { S += red[w2 * 2]; S2 += red[w2 * 2 + 1]; }
    red[0] = S; red[1] = S2;
  }
  __syncthreads();
  const float mean = red[0] * (1.f / (3.f * DD));
  const float var  = red[1] * (1.f / (3.f * DD)) - mean * mean;
  const float inv  = rsqrtf(var + 1e-5f);

  for (int i = tid; i < DD; i += 256) {
    float r = (p[i]          - mean) * inv * ln_g[i]          + ln_b[i];
    float c = (p[i + DD]     - mean) * inv * ln_g[i + DD]     + ln_b[i + DD];
    float u = (p[i + 2 * DD] - mean) * inv * ln_g[i + 2 * DD] + ln_b[i + 2 * DD];
    float sr   = 1.f / (1.f + expf(-r));
    float cand = tanhf(sr * c);
    float up   = 1.f / (1.f + expf(-(u - 1.f)));      // update_bias = -1
    float dold = deter_state[(size_t)b * DD + i];
    float dnew = up * cand + (1.f - up) * dold;
    deter_state[(size_t)b * DD + i] = dnew;
    deters_out[(size_t)b * DD + i]  = dnew;
  }
}

// ---------------------------------------------------------------------------
// Gumbel-max straight-through sample. Forward value of
// probs + sg(onehot - probs) is exactly onehot -> wave32 argmax only.
// One wave per (b,k) row of V=32 logits; 8 rows per block.
// ---------------------------------------------------------------------------
__global__ __launch_bounds__(256) void sample_kernel(
    const float* __restrict__ logits,   // [B*K, V] (posts slice)
    const float* __restrict__ unif,     // [B*K, V] (unif slice)
    float* __restrict__ stoch_out,      // [B, S]   (stochs slice)
    float* __restrict__ stoch_state)    // [B, S]
{
  const int row = blockIdx.x * 8 + (threadIdx.x >> 5);   // b*K + k
  const int v   = threadIdx.x & 31;
  const size_t idxg = (size_t)row * VV_ + v;

  float u = unif[idxg];
  u = fminf(fmaxf(u, 1e-5f), 1.f - 1e-5f);
  float g = -logf(-logf(u));
  float score = logits[idxg] + g;

  int idx = v;
  for (int off = 16; off > 0; off >>= 1) {
    float os = __shfl_down(score, off, 32);
    int   oi = __shfl_down(idx,   off, 32);
    if (os > score || (os == score && oi < idx)) { score = os; idx = oi; }
  }
  idx = __shfl(idx, 0, 32);   // first-max index (jnp.argmax tie-break)

  const float one = (v == idx) ? 1.f : 0.f;
  stoch_out[idxg]   = one;
  stoch_state[idxg] = one;
}

// ---------------------------------------------------------------------------
// Host launcher
// ---------------------------------------------------------------------------
extern "C" void kernel_launch(void* const* d_in, const int* in_sizes, int n_in,
                              void* d_out, int out_size, void* d_ws,
                              size_t ws_size, hipStream_t stream)
{
  const float* obs    = (const float*)d_in[0];
  const float* act    = (const float*)d_in[1];
  const float* deter0 = (const float*)d_in[2];
  const float* stoch0 = (const float*)d_in[3];
  const float* unif   = (const float*)d_in[4];
  const float* W_oo   = (const float*)d_in[5];
  const float* b_oo   = (const float*)d_in[6];
  const float* W_ia   = (const float*)d_in[7];
  const float* b_ia   = (const float*)d_in[8];
  const float* W_is   = (const float*)d_in[9];
  const float* b_is   = (const float*)d_in[10];
  const float* W_gru  = (const float*)d_in[11];
  const float* ln_g   = (const float*)d_in[12];
  const float* ln_b   = (const float*)d_in[13];
  const float* W_od   = (const float*)d_in[14];
  const float* b_od   = (const float*)d_in[15];
  const float* W_op   = (const float*)d_in[16];
  const float* b_op   = (const float*)d_in[17];
  const float* W_io   = (const float*)d_in[18];
  const float* b_io   = (const float*)d_in[19];
  const float* W_ip   = (const float*)d_in[20];
  const float* b_ip   = (const float*)d_in[21];

  float* deters_out = (float*)d_out;                         // [T,B,D]
  float* stochs_out = deters_out + (size_t)TT * BB * DD;     // [T,B,S]
  float* posts_out  = stochs_out + (size_t)TT * BB * SS;     // [T,B,K,V]
  float* priors_out = posts_out  + (size_t)TT * BB * SS;     // [T,B,K,V]

  // workspace carve (256B aligned slabs), ~113 MB total
  char* wp = (char*)d_ws;
  auto carve = [&](size_t bytes) -> char* {
    char* p = wp; wp += (bytes + 255) & ~(size_t)255; return p;
  };
  unsigned short* Wx  = (unsigned short*)carve((size_t)(ACT_ + SS) * HH * 2);
  unsigned short* Wg  = (unsigned short*)carve((size_t)(HH + DD) * 3 * DD * 2);
  unsigned short* Wo  = (unsigned short*)carve((size_t)(DD + OBS_) * HH * 2);
  unsigned short* Wop = (unsigned short*)carve((size_t)HH * SS * 2);
  unsigned short* Wio = (unsigned short*)carve((size_t)DD * HH * 2);
  unsigned short* Wip = (unsigned short*)carve((size_t)HH * SS * 2);
  float* bx   = (float*)carve((size_t)HH * 4);
  float* bo   = (float*)carve((size_t)HH * 4);
  float* Xbuf = (float*)carve((size_t)BB * HH * 4);
  float* Pbuf = (float*)carve((size_t)BB * 3 * DD * 4);
  float* Hbuf = (float*)carve((size_t)BB * HH * 4);
  float* dst  = (float*)carve((size_t)BB * DD * 4);   // deter state
  float* sst  = (float*)carve((size_t)BB * SS * 4);   // stoch state

  // ---- one-time (per launch) bf16 tile-swizzled weight packing ----
  // (bf16 weights ~86MB total -> L2-resident across all 64 steps)
  auto pack = [&](const float* s, unsigned short* d, int N, int Ksrc,
                  int ktOff, int kTilesTot) {
    size_t n = (size_t)Ksrc * N;
    size_t nb = (n + 255) / 256; if (nb > 4096) nb = 4096;
    pack_w_kernel<<<dim3((unsigned)nb), dim3(256), 0, stream>>>(
        s, d, N, Ksrc, ktOff, kTilesTot);
  };
  pack(W_ia,  Wx,  HH,     ACT_, 0,  (ACT_ + SS) / 32);     // kt 0
  pack(W_is,  Wx,  HH,     SS,   1,  (ACT_ + SS) / 32);     // kt 1..32
  pack(W_gru, Wg,  3 * DD, HH + DD, 0, (HH + DD) / 32);
  pack(W_od,  Wo,  HH,     DD,   0,  (DD + OBS_) / 32);     // kt 0..63
  pack(W_oo,  Wo,  HH,     OBS_, DD / 32, (DD + OBS_) / 32);// kt 64..111
  pack(W_op,  Wop, SS,     HH,   0,  HH / 32);
  pack(W_io,  Wio, HH,     DD,   0,  DD / 32);
  pack(W_ip,  Wip, SS,     HH,   0,  HH / 32);
  add2_kernel<<<dim3(HH / 256), dim3(256), 0, stream>>>(b_ia, b_is, bx, HH);
  add2_kernel<<<dim3(HH / 256), dim3(256), 0, stream>>>(b_od, b_oo, bo, HH);

  (void)hipMemcpyAsync(dst, deter0, (size_t)BB * DD * 4,
                       hipMemcpyDeviceToDevice, stream);
  (void)hipMemcpyAsync(sst, stoch0, (size_t)BB * SS * 4,
                       hipMemcpyDeviceToDevice, stream);

  const dim3 blk(256);
  const dim3 gH(HH / BN, BB / BM);          // N=2048 -> 16x8 = 128 blocks
  const dim3 gG(3 * DD / BN, BB / BM);      // N=6144 -> 48x8 = 384 blocks
  const dim3 gS(SS / BN, BB / BM);          // N=1024 -> 8x8  = 64 blocks

  // ---- sequential scan over T ----
  for (int t = 0; t < TT; ++t) {
    // x = elu([act_t | stoch] @ [W_ia;W_is] + (b_ia+b_is))
    gemm_bf16_wmma<<<gH, blk, 0, stream>>>(
        act + (size_t)t * BB * ACT_, ACT_, ACT_, sst, SS,
        Wx, HH, ACT_ + SS, bx, Xbuf, HH, 1);
    // parts_raw = [x | deter] @ W_gru
    gemm_bf16_wmma<<<gG, blk, 0, stream>>>(
        Xbuf, HH, HH, dst, DD,
        Wg, 3 * DD, HH + DD, nullptr, Pbuf, 3 * DD, 0);
    // layernorm + GRU -> deter_t (state + output)
    gru_ln_kernel<<<dim3(BB), blk, 0, stream>>>(
        Pbuf, ln_g, ln_b, dst, deters_out + (size_t)t * BB * DD);
    // h = elu([deter_t | obs_t] @ [W_od;W_oo] + (b_od+b_oo))
    gemm_bf16_wmma<<<gH, blk, 0, stream>>>(
        dst, DD, DD, obs + (size_t)t * BB * OBS_, OBS_,
        Wo, HH, DD + OBS_, bo, Hbuf, HH, 1);
    // posts_t (logits) = h @ W_op + b_op  -> straight into d_out
    gemm_bf16_wmma<<<gS, blk, 0, stream>>>(
        Hbuf, HH, HH, nullptr, 0,
        Wop, SS, HH, b_op, posts_out + (size_t)t * BB * SS, SS, 0);
    // stoch_t = onehot(argmax(logits + gumbel))
    sample_kernel<<<dim3(BB * KK_ / 8), blk, 0, stream>>>(
        posts_out + (size_t)t * BB * SS,
        unif + (size_t)t * BB * SS,
        stochs_out + (size_t)t * BB * SS, sst);
  }

  // ---- priors over all deters ----
  for (int t = 0; t < TT; ++t) {
    gemm_bf16_wmma<<<gH, blk, 0, stream>>>(
        deters_out + (size_t)t * BB * DD, DD, DD, nullptr, 0,
        Wio, HH, DD, b_io, Hbuf, HH, 1);
    gemm_bf16_wmma<<<gS, blk, 0, stream>>>(
        Hbuf, HH, HH, nullptr, 0,
        Wip, SS, HH, b_ip, priors_out + (size_t)t * BB * SS, SS, 0);
  }
  (void)in_sizes; (void)n_in; (void)out_size; (void)ws_size;
}
